// EncoderBlock_87754771792403
// MI455X (gfx1250) — matrix operand order repaired and checked by reading, not compiled, not run on hardware
//
#include <hip/hip_runtime.h>
#include <math.h>

// ---------------- types ----------------
typedef __bf16 bf16;
typedef bf16  v16bf __attribute__((ext_vector_type(16)));
typedef float v8f   __attribute__((ext_vector_type(8)));

#define WMMA_BF16(a, b, c) \
  __builtin_amdgcn_wmma_f32_16x16x32_bf16(false, (a), false, (b), (short)0, (c), false, false)

__device__ __forceinline__ v8f zero8() {
  v8f z;
#pragma unroll
  for (int i = 0; i < 8; ++i) z[i] = 0.0f;
  return z;
}

// A fragment (16x32 bf16, MxK): lane L(0..15)=row, holds K = hs*16 + [0..7] and hs*16+8 + [0..7]
// -> two contiguous 16-byte chunks.
__device__ __forceinline__ v16bf load_a_frag(const bf16* p0, const bf16* p1) {
  union { v16bf v; uint4 q[2]; } u;
  u.q[0] = *(const uint4*)p0;
  u.q[1] = *(const uint4*)p1;
  return u.v;
}
// B fragment (32x16 bf16, KxN): lane L(0..15)=col holds K=0..15, lane L+16 holds K=16..31
// -> one contiguous 32-byte chunk per lane.
__device__ __forceinline__ v16bf load_b_frag(const bf16* p) {
  union { v16bf v; uint4 q[2]; } u;
  u.q[0] = *(const uint4*)p;
  u.q[1] = *(const uint4*)(p + 8);
  return u.v;
}

// ---------------- fp32 -> bf16 convert ----------------
__global__ void cvt_bf16_kernel(const float* __restrict__ src, bf16* __restrict__ dst, int n) {
  int i = blockIdx.x * blockDim.x + threadIdx.x;
  if (i < n) dst[i] = (bf16)src[i];
}

// ---------------- WMMA GEMM (compile-time segment / mode) ----------------
// out[m, n] = act( A[m,:] . W[n,:] + bias[n] ) (+ resid)  with segment row mapping:
//   flat_token(m) = (m / SEG_LEN) * 1024 + SEG_OFF + (m % SEG_LEN)
// A: bf16 [Mtot, 512], W: bf16 [512, 512] row-major (N x K).
// Block tile 128x128 (8 waves as 4Mx2N), wave tile 32x64 (2 M-subtiles x 4 N-subtiles)
// -> 8 WMMAs per 6 fragment loads in the K loop (A reused 4x, B reused 2x).
// MODE 0: fp32 out [flat,512] (+resid[flat,512])
// MODE 1: bf16 out [flat,512] (optional relu)
// MODE 2: bf16 out Q/K layout  [(b*8+h)*1024 + t]*64 + d
// MODE 3: bf16 out V^T layout  [(b*8+h)*64 + d]*1024 + t
template <int SEG_LEN, int MODE, int RELU>
__global__ void __launch_bounds__(256)
gemm_wmma_kernel(const bf16* __restrict__ A, const bf16* __restrict__ W,
                 const float* __restrict__ bias, const float* __restrict__ resid,
                 float* __restrict__ outF, bf16* __restrict__ outB) {
  constexpr int SEG_OFF = (SEG_LEN == 768) ? 256 : 0;

  const int lane = threadIdx.x & 31;
  const int wv   = threadIdx.x >> 5;   // 0..7
  const int msub = wv & 3;             // 4 waves along M (32 rows each)
  const int nsub = wv >> 2;            // 2 waves along N (64 cols each)
  const int r    = lane & 15;
  const int hs   = lane >> 4;

  const int mbase = blockIdx.y * 128 + msub * 32;
  const int nbase = blockIdx.x * 128 + nsub * 64;

  const int am0 = mbase + r;
  const int am1 = am0 + 16;
  // compile-time SEG_LEN -> shifts / magic multiplies, no runtime div sequence
  const long fa0 = (long)(am0 / SEG_LEN) * 1024 + SEG_OFF + (am0 % SEG_LEN);
  const long fa1 = (long)(am1 / SEG_LEN) * 1024 + SEG_OFF + (am1 % SEG_LEN);
  const bf16* a0p = A + fa0 * 512;
  const bf16* a1p = A + fa1 * 512;
  const bf16* bp[4];
#pragma unroll
  for (int j = 0; j < 4; ++j) bp[j] = W + (long)(nbase + j * 16 + r) * 512;

  v8f acc0[4], acc1[4];
#pragma unroll
  for (int j = 0; j < 4; ++j) { acc0[j] = zero8(); acc1[j] = zero8(); }

#pragma unroll 2
  for (int k = 0; k < 512; k += 32) {
    if (k + 64 < 512) {   // uniform; speculative prefetch -> global_prefetch_b8
      __builtin_prefetch(a0p + k + 64, 0, 1);
      __builtin_prefetch(bp[0] + k + 64, 0, 1);
    }
    v16bf a0 = load_a_frag(a0p + k + hs * 16, a0p + k + 8 + hs * 16);
    v16bf a1 = load_a_frag(a1p + k + hs * 16, a1p + k + 8 + hs * 16);
#pragma unroll
    for (int j = 0; j < 4; ++j) {
      v16bf b = load_b_frag(bp[j] + k + hs * 16);
      acc0[j] = WMMA_BF16(a0, b, acc0[j]);
      acc1[j] = WMMA_BF16(a1, b, acc1[j]);
    }
  }

  float biasj[4];
#pragma unroll
  for (int j = 0; j < 4; ++j) biasj[j] = bias ? bias[nbase + j * 16 + r] : 0.0f;

#pragma unroll
  for (int v = 0; v < 8; ++v) {
    const int mrow0 = mbase + v + 8 * hs;   // C layout: VGPR v -> row v / v+8
    const int mrow1 = mrow0 + 16;
#pragma unroll
    for (int j = 0; j < 4; ++j) {
      const int c = nbase + j * 16 + r;
#pragma unroll
      for (int i = 0; i < 2; ++i) {
        float val = (i == 0 ? acc0[j][v] : acc1[j][v]) + biasj[j];
        const int m = (i == 0) ? mrow0 : mrow1;
        const int b = m / SEG_LEN;                 // compile-time strength-reduced
        const int t = SEG_OFF + (m % SEG_LEN);
        const long flat = (long)b * 1024 + t;
        if constexpr (MODE == 0) {
          if (resid) val += resid[flat * 512 + c];
          outF[flat * 512 + c] = val;
        } else if constexpr (MODE == 1) {
          if constexpr (RELU) { if (val < 0.0f) val = 0.0f; }
          outB[flat * 512 + c] = (bf16)val;
        } else if constexpr (MODE == 2) {
          const int hh = c >> 6, d = c & 63;
          outB[(((long)(b * 8 + hh) * 1024 + t) << 6) + d] = (bf16)val;
        } else {
          const int hh = c >> 6, d = c & 63;
          outB[((long)(b * 8 + hh) * 64 + d) * 1024 + t] = (bf16)val;
        }
      }
    }
  }
}

// ---------------- attention (per head, WMMA scores + softmax-in-LDS + WMMA PV) ----------------
// qb,kb: bf16 [(b*8+h)*1024 + tok]*64 + d ; vT: bf16 [(b*8+h)*64 + d]*1024 + tok
// zb:    bf16 [b*1024 + tok]*512 + h*64 + d
template <int SEG_LEN>
__global__ void __launch_bounds__(64)
attn_wmma_kernel(const bf16* __restrict__ qb, const bf16* __restrict__ kb,
                 const bf16* __restrict__ vT, bf16* __restrict__ zb) {
  constexpr int SEG_OFF = (SEG_LEN == 768) ? 256 : 0;
  extern __shared__ char smem[];
  const int w    = threadIdx.x >> 5;   // wave 0/1
  const int lane = threadIdx.x & 31;
  const int r    = lane & 15;
  const int hs   = lane >> 4;
  const int bh   = blockIdx.y;         // 0..127
  const int b    = bh >> 3, h = bh & 7;
  const int qbase = blockIdx.x * 32 + w * 16;   // within segment

  float* scores = (float*)smem + w * 16 * SEG_LEN;
  bf16*  pmat   = (bf16*)(smem + (size_t)2 * 16 * SEG_LEN * 4) + w * 16 * SEG_LEN;
  float* rowsum = (float*)(smem + (size_t)2 * 16 * SEG_LEN * 4 + (size_t)2 * 16 * SEG_LEN * 2) + w * 16;

  const long headBase = (long)bh * 1024;

  // Q fragments for this wave's 16 rows (dk = 64 -> two K=32 fragments)
  const bf16* qrow = qb + (headBase + SEG_OFF + qbase + r) * 64;
  const v16bf qa0 = load_a_frag(qrow + hs * 16,      qrow + 8 + hs * 16);
  const v16bf qa1 = load_a_frag(qrow + 32 + hs * 16, qrow + 40 + hs * 16);

  // ---- scores S = Q K^T / 8 ----
#pragma unroll 4
  for (int n0 = 0; n0 < SEG_LEN; n0 += 16) {
    const bf16* krow = kb + (headBase + SEG_OFF + n0 + r) * 64;
    v16bf kb0 = load_b_frag(krow + hs * 16);
    v16bf kb1 = load_b_frag(krow + 32 + hs * 16);
    v8f acc = zero8();
    acc = WMMA_BF16(qa0, kb0, acc);
    acc = WMMA_BF16(qa1, kb1, acc);
#pragma unroll
    for (int v = 0; v < 8; ++v)
      scores[(v + 8 * hs) * SEG_LEN + n0 + r] = acc[v] * 0.125f;
  }
  __syncthreads();

  // ---- softmax: 2 lanes per row (row = r, half = hs) ----
  {
    constexpr int HALF_N = SEG_LEN >> 1;
    float* srow = scores + r * SEG_LEN + hs * HALF_N;
    float mx = -3.4e38f;
#pragma unroll 4
    for (int i = 0; i < HALF_N; ++i) mx = fmaxf(mx, srow[i]);
    mx = fmaxf(mx, __shfl_xor(mx, 16, 32));
    bf16* prow = pmat + r * SEG_LEN + hs * HALF_N;
    float sum = 0.0f;
#pragma unroll 4
    for (int i = 0; i < HALF_N; ++i) {
      float e = __expf(srow[i] - mx);
      sum += e;
      prow[i] = (bf16)e;
    }
    sum += __shfl_xor(sum, 16, 32);
    if (hs == 0) rowsum[r] = sum;
  }
  __syncthreads();

  // ---- O = P V ----
  v8f o0 = zero8(), o1 = zero8(), o2 = zero8(), o3 = zero8();
#pragma unroll 2
  for (int k0 = 0; k0 < SEG_LEN; k0 += 32) {
    const bf16* pr = pmat + r * SEG_LEN + k0;
    v16bf pa = load_a_frag(pr + hs * 16, pr + 8 + hs * 16);
    const bf16* vbase = vT + (long)bh * 64 * 1024 + SEG_OFF + k0 + hs * 16;
    v16bf vb0 = load_b_frag(vbase + (0 * 16 + r) * 1024);
    v16bf vb1 = load_b_frag(vbase + (1 * 16 + r) * 1024);
    v16bf vb2 = load_b_frag(vbase + (2 * 16 + r) * 1024);
    v16bf vb3 = load_b_frag(vbase + (3 * 16 + r) * 1024);
    o0 = WMMA_BF16(pa, vb0, o0);
    o1 = WMMA_BF16(pa, vb1, o1);
    o2 = WMMA_BF16(pa, vb2, o2);
    o3 = WMMA_BF16(pa, vb3, o3);
  }

#pragma unroll
  for (int v = 0; v < 8; ++v) {
    const int m = v + 8 * hs;
    const float inv = 1.0f / rowsum[m];
    const int tq = SEG_OFF + qbase + m;
    const long obase = ((long)b * 1024 + tq) * 512 + h * 64;
    zb[obase + 0 * 16 + r] = (bf16)(o0[v] * inv);
    zb[obase + 1 * 16 + r] = (bf16)(o1[v] * inv);
    zb[obase + 2 * 16 + r] = (bf16)(o2[v] * inv);
    zb[obase + 3 * 16 + r] = (bf16)(o3[v] * inv);
  }
}

// ---------------- BatchNorm over (B, N) ----------------
__global__ void zero_f32_kernel(float* p, int n) {
  int i = blockIdx.x * blockDim.x + threadIdx.x;
  if (i < n) p[i] = 0.0f;
}
__global__ void bn_stats_kernel(const float* __restrict__ src, float* __restrict__ stats) {
  const int c  = blockIdx.x * blockDim.x + threadIdx.x;   // channel 0..511
  const long r0 = (long)blockIdx.y * 256;
  float s = 0.0f, s2 = 0.0f;
#pragma unroll 8
  for (int i = 0; i < 256; ++i) {
    float v = src[(r0 + i) * 512 + c];
    s += v; s2 += v * v;
  }
  atomicAdd(&stats[c], s);
  atomicAdd(&stats[512 + c], s2);
}
__global__ void bn_apply_kernel(const float* __restrict__ src, const float* __restrict__ stats,
                                const float* __restrict__ g, const float* __restrict__ bta,
                                float* __restrict__ dstF, bf16* __restrict__ dstB) {
  const long i = (long)blockIdx.x * blockDim.x + threadIdx.x;
  const int c = (int)(i & 511);
  const float mean = stats[c] * (1.0f / 16384.0f);
  const float var  = stats[512 + c] * (1.0f / 16384.0f) - mean * mean;
  const float y = (src[i] - mean) * rsqrtf(var + 1e-5f) * g[c] + bta[c];
  dstF[i] = y;
  if (dstB) dstB[i] = (bf16)y;
}

// ---------------- launch ----------------
extern "C" void kernel_launch(void* const* d_in, const int* in_sizes, int n_in,
                              void* d_out, int out_size, void* d_ws, size_t ws_size,
                              hipStream_t stream) {
  (void)in_sizes; (void)n_in; (void)out_size; (void)ws_size;

  const long TOK = 16L * 1024;   // 16384 tokens
  const long E   = 512;

  const float* x = (const float*)d_in[0];
  // weight order: rq rk rv ro tq tk tv to f1 f2 (w at odd idx 1..19, b at idx+1)
  const float* Wsrc[10]; const float* Bsrc[10];
  for (int i = 0; i < 10; ++i) { Wsrc[i] = (const float*)d_in[1 + 2 * i]; Bsrc[i] = (const float*)d_in[2 + 2 * i]; }
  const float* bn1_g = (const float*)d_in[21];
  const float* bn1_b = (const float*)d_in[22];
  const float* bn2_g = (const float*)d_in[23];
  const float* bn2_b = (const float*)d_in[24];
  float* out = (float*)d_out;

  // workspace layout
  char* base = (char*)d_ws;
  size_t off = 0;
  bf16* xbf  = (bf16*)(base + off); off += (size_t)TOK * E * 2;        // 16.8 MB
  bf16* wbf  = (bf16*)(base + off); off += (size_t)10 * E * E * 2;     //  5.2 MB
  bf16* qbuf = (bf16*)(base + off); off += (size_t)TOK * E * 2;        // [bh, tok, d]
  bf16* kbuf = (bf16*)(base + off); off += (size_t)TOK * E * 2;
  bf16* vTb  = (bf16*)(base + off); off += (size_t)TOK * E * 2;        // [bh, d, tok]
  bf16* zbf  = (bf16*)(base + off); off += (size_t)TOK * E * 2;
  float* h   = (float*)(base + off); off += (size_t)TOK * E * 4;       // 33.5 MB (reused as FFN out)
  float* hn  = (float*)(base + off); off += (size_t)TOK * E * 4;
  bf16* hnb  = (bf16*)(base + off); off += (size_t)TOK * E * 2;
  bf16* h1b  = (bf16*)(base + off); off += (size_t)TOK * E * 2;
  float* stats = (float*)(base + off); off += 1024 * 4;

  // --- convert x and weights to bf16 ---
  cvt_bf16_kernel<<<(int)((TOK * E) / 256), 256, 0, stream>>>(x, xbf, (int)(TOK * E));
  for (int i = 0; i < 10; ++i)
    cvt_bf16_kernel<<<(int)((E * E) / 256), 256, 0, stream>>>(Wsrc[i], wbf + (size_t)i * E * E, (int)(E * E));

  constexpr int SEG_R = 256, SEG_T = 768;
  const dim3 gR(4, (16 * SEG_R) / 128);   // (4, 32)
  const dim3 gT(4, (16 * SEG_T) / 128);   // (4, 96)
  const dim3 gF(4, (int)(TOK / 128));     // (4, 128)

  // --- QKV projections (robot / task), heads layout + V transposed ---
  gemm_wmma_kernel<SEG_R, 2, 0><<<gR, 256, 0, stream>>>(xbf, wbf + 0 * E * E, Bsrc[0], nullptr, nullptr, qbuf);
  gemm_wmma_kernel<SEG_R, 2, 0><<<gR, 256, 0, stream>>>(xbf, wbf + 1 * E * E, Bsrc[1], nullptr, nullptr, kbuf);
  gemm_wmma_kernel<SEG_R, 3, 0><<<gR, 256, 0, stream>>>(xbf, wbf + 2 * E * E, Bsrc[2], nullptr, nullptr, vTb);
  gemm_wmma_kernel<SEG_T, 2, 0><<<gT, 256, 0, stream>>>(xbf, wbf + 4 * E * E, Bsrc[4], nullptr, nullptr, qbuf);
  gemm_wmma_kernel<SEG_T, 2, 0><<<gT, 256, 0, stream>>>(xbf, wbf + 5 * E * E, Bsrc[5], nullptr, nullptr, kbuf);
  gemm_wmma_kernel<SEG_T, 3, 0><<<gT, 256, 0, stream>>>(xbf, wbf + 6 * E * E, Bsrc[6], nullptr, nullptr, vTb);

  // --- attention per head (2 waves / block, dynamic LDS for scores + P) ---
  const size_t smemR = (size_t)2 * 16 * SEG_R * 4 + (size_t)2 * 16 * SEG_R * 2 + 2 * 16 * 4;
  const size_t smemT = (size_t)2 * 16 * SEG_T * 4 + (size_t)2 * 16 * SEG_T * 2 + 2 * 16 * 4;
  attn_wmma_kernel<SEG_R><<<dim3(SEG_R / 32, 128), 64, smemR, stream>>>(qbuf, kbuf, vTb, zbf);
  attn_wmma_kernel<SEG_T><<<dim3(SEG_T / 32, 128), 64, smemT, stream>>>(qbuf, kbuf, vTb, zbf);

  // --- output projection + residual x -> h (fp32) ---
  gemm_wmma_kernel<SEG_R, 0, 0><<<gR, 256, 0, stream>>>(zbf, wbf + 3 * E * E, Bsrc[3], x, h, nullptr);
  gemm_wmma_kernel<SEG_T, 0, 0><<<gT, 256, 0, stream>>>(zbf, wbf + 7 * E * E, Bsrc[7], x, h, nullptr);

  // --- BN1 -> hn (fp32) + hnb (bf16) ---
  zero_f32_kernel<<<4, 256, 0, stream>>>(stats, 1024);
  bn_stats_kernel<<<dim3(2, 64), 256, 0, stream>>>(h, stats);
  bn_apply_kernel<<<(int)((TOK * E) / 256), 256, 0, stream>>>(h, stats, bn1_g, bn1_b, hn, hnb);

  // --- FFN: h1 = relu(hn @ f1^T + b) (bf16); g = h1 @ f2^T + b + hn (fp32, reuse h) ---
  gemm_wmma_kernel<1024, 1, 1><<<gF, 256, 0, stream>>>(hnb, wbf + 8 * E * E, Bsrc[8], nullptr, nullptr, h1b);
  gemm_wmma_kernel<1024, 0, 0><<<gF, 256, 0, stream>>>(h1b, wbf + 9 * E * E, Bsrc[9], hn, h, nullptr);

  // --- BN2 -> d_out ---
  zero_f32_kernel<<<4, 256, 0, stream>>>(stats, 1024);
  bn_stats_kernel<<<dim3(2, 64), 256, 0, stream>>>(h, stats);
  bn_apply_kernel<<<(int)((TOK * E) / 256), 256, 0, stream>>>(h, stats, bn2_g, bn2_b, out, nullptr);
}
